// LlamaQMLP_47751446397246
// MI455X (gfx1250) — matrix operand order repaired and checked by reading, not compiled, not run on hardware
//
#include <hip/hip_runtime.h>
#include <hip/hip_bf16.h>

// ---------------------------------------------------------------- constants
static constexpr int HID = 4096;    // hidden
static constexpr int IMD = 11008;   // intermediate
static constexpr int RNK = 32;      // low-rank
static constexpr int MTOK = 4096;   // B*S tokens
static constexpr float Q_ZERO = 7.5f;
static constexpr float Q_STEP = 0.005f;
static constexpr float LR_SCALE = 0.02f;

typedef __attribute__((ext_vector_type(16))) _Float16 v16h;
typedef __attribute__((ext_vector_type(8)))  float    v8f;
typedef __attribute__((ext_vector_type(4)))  _Float16 v4h;
typedef __attribute__((ext_vector_type(2)))  _Float16 v2h;
typedef __attribute__((ext_vector_type(4)))  int      v4i;

// gfx1250 async global->LDS copy path (ASYNCcnt), if the toolchain exposes it.
#if defined(__has_builtin)
#if __has_builtin(__builtin_amdgcn_global_load_async_to_lds_b128) && \
    __has_builtin(__builtin_amdgcn_s_wait_asynccnt)
#define ASYNC_LDS 1
#endif
#endif
#ifndef ASYNC_LDS
#define ASYNC_LDS 0
#endif

#if ASYNC_LDS
#define GLOBAL_AS __attribute__((address_space(1)))
#define LDS_AS    __attribute__((address_space(3)))
__device__ __forceinline__ void async_cp16(const _Float16* g, _Float16* l) {
    // builtin signature (per hipcc diagnostic): (v4i* src, v4i* lds_dst, imm, imm)
    __builtin_amdgcn_global_load_async_to_lds_b128(
        (GLOBAL_AS v4i*)g, (LDS_AS v4i*)l, 0, 0);
}
#endif

// Pack (q - 7.5) for 4 int4 levels into 4 f16 (exact: multiples of 0.5).
__device__ __forceinline__ v4h dq4(int4 q) {
    union { v4h v; v2h h[2]; } u;
    u.h[0] = __builtin_bit_cast(v2h,
        __builtin_amdgcn_cvt_pkrtz((float)q.x - Q_ZERO, (float)q.y - Q_ZERO));
    u.h[1] = __builtin_bit_cast(v2h,
        __builtin_amdgcn_cvt_pkrtz((float)q.z - Q_ZERO, (float)q.w - Q_ZERO));
    return u.v;
}

// Load a 32-byte (16 x f16) fragment from two 16-byte LDS chunks.
__device__ __forceinline__ v16h ld_frag16(const _Float16* p0, const _Float16* p1) {
    union { v16h v; float4 f[2]; } u;
    u.f[0] = *(const float4*)p0;
    u.f[1] = *(const float4*)p1;
    return u.v;
}

// A fragment: 16(M) x 32(K) f16, LDS tile stored [m][k].
__device__ __forceinline__ v16h ld_a_frag(const _Float16* As, int mbase, int lane) {
    int row = mbase + (lane & 15);
    int kh  = (lane >> 4) * 8;
    const _Float16* p = As + row * 32;
    return ld_frag16(p + kh, p + 16 + kh);
}

// B fragment: 32(K) x 16(N) f16, LDS tile stored [n][k].
__device__ __forceinline__ v16h ld_b_frag(const _Float16* Bs, int nbase, int lane) {
    int col = nbase + (lane & 15);
    int kc  = (lane >> 4) * 8;
    const _Float16* p = Bs + col * 32 + kc;
    return ld_frag16(p, p + 16);
}

__device__ __forceinline__ v8f wmma_f16(v16h a, v16h b, v8f c) {
    return __builtin_amdgcn_wmma_f32_16x16x32_f16(false, a, false, b, (short)0, c, false, false);
}

// ---------------------------------------------------------------- prep:
// xh = f16(x);  Pg = x @ gate_A;  Pu = x @ up_A   (one block per token row)
__global__ __launch_bounds__(256) void prep_kernel(
    const float* __restrict__ x, const float* __restrict__ gA,
    const float* __restrict__ uA, _Float16* __restrict__ xh,
    float* __restrict__ Pg, float* __restrict__ Pu)
{
    const int m = blockIdx.x;
    const float* xr = x + (size_t)m * HID;
    _Float16* xo = xh + (size_t)m * HID;

    for (int t = threadIdx.x; t < HID / 4; t += 256) {
        float4 v = ((const float4*)xr)[t];
        v4h h;
        h[0] = (_Float16)v.x; h[1] = (_Float16)v.y;
        h[2] = (_Float16)v.z; h[3] = (_Float16)v.w;
        ((v4h*)xo)[t] = h;
    }

    const int r   = threadIdx.x & 31;
    const int seg = threadIdx.x >> 5;
    float ag = 0.f, au = 0.f;
    const int h0 = seg * (HID / 8), h1 = h0 + (HID / 8);
    for (int h = h0; h < h1; ++h) {
        float xv = xr[h];
        ag += xv * gA[(size_t)h * RNK + r];
        au += xv * uA[(size_t)h * RNK + r];
    }
    __shared__ float red[2][8][32];
    red[0][seg][r] = ag;
    red[1][seg][r] = au;
    __syncthreads();
    if (threadIdx.x < 32) {
        float s = 0.f;
        for (int k = 0; k < 8; ++k) s += red[0][k][threadIdx.x];
        Pg[(size_t)m * RNK + threadIdx.x] = s;
    } else if (threadIdx.x < 64) {
        int rr = threadIdx.x - 32;
        float s = 0.f;
        for (int k = 0; k < 8; ++k) s += red[1][k][rr];
        Pu[(size_t)m * RNK + rr] = s;
    }
}

// ---------------------------------------------------------------- fused
// gate+up GEMM, double-buffered software pipeline, SwiGLU epilogue -> f16 h.
// Block tile 128x128, 8 waves (2x4), each wave 64x32 via 4x2 WMMA frags.
__global__ __launch_bounds__(256) void gateup_kernel(
    const _Float16* __restrict__ xh,
    const int* __restrict__ gq, const int* __restrict__ uq,
    const float* __restrict__ gscale, const float* __restrict__ uscale,
    const float* __restrict__ gB, const float* __restrict__ uB,
    const float* __restrict__ Pg, const float* __restrict__ Pu,
    _Float16* __restrict__ hbuf)
{
    __shared__ __align__(16) _Float16 As[2][128 * 32];
    __shared__ __align__(16) _Float16 Bg[2][128 * 32];
    __shared__ __align__(16) _Float16 Bu[2][128 * 32];

    const int nblk = blockIdx.x, mblk = blockIdx.y;
    const int tid = threadIdx.x;
    const int lane = tid & 31, wave = tid >> 5;
    const int wm = wave >> 2, wn = wave & 3;

    v8f cg[4][2] = {}, cu[4][2] = {};

    const size_t arow0 = (size_t)(mblk * 128) * HID;
    const size_t grow0 = (size_t)(nblk * 128) * HID;

    int4 qg[4], qu[4];
#if !ASYNC_LDS
    float4 av[2];
#endif

    auto fill_load = [&](int k0, int buf) {
        #pragma unroll
        for (int s = 0; s < 2; ++s) {
            int c = tid * 2 + s;
            int row = c >> 2, cof = (c & 3) * 8;
            const _Float16* gp = xh + arow0 + (size_t)row * HID + k0 + cof;
#if ASYNC_LDS
            async_cp16(gp, &As[buf][row * 32 + cof]);
#else
            av[s] = *(const float4*)gp;
#endif
        }
        #pragma unroll
        for (int s = 0; s < 4; ++s) {
            int c = tid * 4 + s;
            int row = c >> 3, cof = (c & 7) * 4;
            const size_t off = grow0 + (size_t)row * HID + k0 + cof;
            qg[s] = *(const int4*)(gq + off);
            qu[s] = *(const int4*)(uq + off);
            if (s == 0 && k0 + 32 < HID) {   // L2 prefetch: one K-tile ahead
                __builtin_prefetch(gq + off + 32, 0, 1);
                __builtin_prefetch(uq + off + 32, 0, 1);
            }
        }
    };
    auto fill_store = [&](int buf) {
        #pragma unroll
        for (int s = 0; s < 4; ++s) {
            int c = tid * 4 + s;
            int row = c >> 3, cof = (c & 7) * 4;
            *(v4h*)(&Bg[buf][row * 32 + cof]) = dq4(qg[s]);
            *(v4h*)(&Bu[buf][row * 32 + cof]) = dq4(qu[s]);
        }
#if !ASYNC_LDS
        #pragma unroll
        for (int s = 0; s < 2; ++s) {
            int c = tid * 2 + s;
            int row = c >> 2, cof = (c & 3) * 8;
            *(float4*)(&As[buf][row * 32 + cof]) = av[s];
        }
#endif
    };

    fill_load(0, 0);
    fill_store(0);

    const int NIT = HID / 32;
    for (int it = 0; it < NIT; ++it) {
        const int buf = it & 1;
#if ASYNC_LDS
        __builtin_amdgcn_s_wait_asynccnt(0);
#endif
        __syncthreads();
        if (it + 1 < NIT) fill_load((it + 1) * 32, buf ^ 1);  // overlap w/ WMMAs

        v16h bg0 = ld_b_frag(&Bg[buf][0], wn * 32,      lane);
        v16h bg1 = ld_b_frag(&Bg[buf][0], wn * 32 + 16, lane);
        v16h bu0 = ld_b_frag(&Bu[buf][0], wn * 32,      lane);
        v16h bu1 = ld_b_frag(&Bu[buf][0], wn * 32 + 16, lane);
        #pragma unroll
        for (int i = 0; i < 4; ++i) {
            v16h a = ld_a_frag(&As[buf][0], wm * 64 + i * 16, lane);
            cg[i][0] = wmma_f16(a, bg0, cg[i][0]);
            cg[i][1] = wmma_f16(a, bg1, cg[i][1]);
            cu[i][0] = wmma_f16(a, bu0, cu[i][0]);
            cu[i][1] = wmma_f16(a, bu1, cu[i][1]);
        }
        if (it + 1 < NIT) fill_store(buf ^ 1);                // cvt after WMMAs
    }

    // ---- epilogue: per-channel scale, low-rank correction, SwiGLU, f16 out
    #pragma unroll
    for (int i = 0; i < 4; ++i) {
        #pragma unroll
        for (int j = 0; j < 2; ++j) {
            const int mb = mblk * 128 + wm * 64 + i * 16 + ((lane >> 4) << 3);
            const int n  = nblk * 128 + wn * 32 + j * 16 + (lane & 15);
            const float sg = gscale[n] * Q_STEP;
            const float su = uscale[n] * Q_STEP;
            #pragma unroll
            for (int e = 0; e < 8; ++e) {
                const int m = mb + e;
                float g = cg[i][j][e] * sg;
                float u = cu[i][j][e] * su;
                const float* pg = Pg + (size_t)m * RNK;
                const float* pu = Pu + (size_t)m * RNK;
                float lg = 0.f, lu = 0.f;
                for (int r = 0; r < RNK; ++r) {
                    lg += pg[r] * gB[(size_t)r * IMD + n];
                    lu += pu[r] * uB[(size_t)r * IMD + n];
                }
                g += LR_SCALE * lg;
                u += LR_SCALE * lu;
                const float silu = g / (1.0f + __expf(-g));
                hbuf[(size_t)m * IMD + n] = (_Float16)(silu * u);
            }
        }
    }
}

// ---------------------------------------------------------------- Ph = h @ down_A
__global__ __launch_bounds__(256) void projh_kernel(
    const _Float16* __restrict__ hbuf, const float* __restrict__ dA,
    float* __restrict__ Ph)
{
    const int m = blockIdx.x;
    const _Float16* hr = hbuf + (size_t)m * IMD;
    const int r   = threadIdx.x & 31;
    const int seg = threadIdx.x >> 5;
    float acc = 0.f;
    const int i0 = seg * (IMD / 8), i1 = i0 + (IMD / 8);
    for (int i = i0; i < i1; ++i)
        acc += (float)hr[i] * dA[(size_t)i * RNK + r];
    __shared__ float red[8][32];
    red[seg][r] = acc;
    __syncthreads();
    if (threadIdx.x < 32) {
        float s = 0.f;
        for (int k = 0; k < 8; ++k) s += red[k][threadIdx.x];
        Ph[(size_t)m * RNK + threadIdx.x] = s;
    }
}

// ---------------------------------------------------------------- down GEMM:
// out = h @ dq(down)^T + LR;  K = IMD, N = HID, f32 output.
__global__ __launch_bounds__(256) void down_kernel(
    const _Float16* __restrict__ hbuf, const int* __restrict__ dq,
    const float* __restrict__ dscale, const float* __restrict__ dB,
    const float* __restrict__ Ph, float* __restrict__ out)
{
    __shared__ __align__(16) _Float16 As[2][128 * 32];
    __shared__ __align__(16) _Float16 Bs[2][128 * 32];

    const int nblk = blockIdx.x, mblk = blockIdx.y;
    const int tid = threadIdx.x;
    const int lane = tid & 31, wave = tid >> 5;
    const int wm = wave >> 2, wn = wave & 3;

    v8f cacc[4][2] = {};

    const size_t arow0 = (size_t)(mblk * 128) * IMD;
    const size_t wrow0 = (size_t)(nblk * 128) * IMD;

    int4 qd[4];
#if !ASYNC_LDS
    float4 av[2];
#endif

    auto fill_load = [&](int k0, int buf) {
        #pragma unroll
        for (int s = 0; s < 2; ++s) {
            int c = tid * 2 + s;
            int row = c >> 2, cof = (c & 3) * 8;
            const _Float16* gp = hbuf + arow0 + (size_t)row * IMD + k0 + cof;
#if ASYNC_LDS
            async_cp16(gp, &As[buf][row * 32 + cof]);
#else
            av[s] = *(const float4*)gp;
#endif
        }
        #pragma unroll
        for (int s = 0; s < 4; ++s) {
            int c = tid * 4 + s;
            int row = c >> 3, cof = (c & 7) * 4;
            const size_t off = wrow0 + (size_t)row * IMD + k0 + cof;
            qd[s] = *(const int4*)(dq + off);
            if (s == 0 && k0 + 32 < IMD)
                __builtin_prefetch(dq + off + 32, 0, 1);
        }
    };
    auto fill_store = [&](int buf) {
        #pragma unroll
        for (int s = 0; s < 4; ++s) {
            int c = tid * 4 + s;
            int row = c >> 3, cof = (c & 7) * 4;
            *(v4h*)(&Bs[buf][row * 32 + cof]) = dq4(qd[s]);
        }
#if !ASYNC_LDS
        #pragma unroll
        for (int s = 0; s < 2; ++s) {
            int c = tid * 2 + s;
            int row = c >> 2, cof = (c & 3) * 8;
            *(float4*)(&As[buf][row * 32 + cof]) = av[s];
        }
#endif
    };

    fill_load(0, 0);
    fill_store(0);

    const int NIT = IMD / 32;
    for (int it = 0; it < NIT; ++it) {
        const int buf = it & 1;
#if ASYNC_LDS
        __builtin_amdgcn_s_wait_asynccnt(0);
#endif
        __syncthreads();
        if (it + 1 < NIT) fill_load((it + 1) * 32, buf ^ 1);

        v16h b0 = ld_b_frag(&Bs[buf][0], wn * 32,      lane);
        v16h b1 = ld_b_frag(&Bs[buf][0], wn * 32 + 16, lane);
        #pragma unroll
        for (int i = 0; i < 4; ++i) {
            v16h a = ld_a_frag(&As[buf][0], wm * 64 + i * 16, lane);
            cacc[i][0] = wmma_f16(a, b0, cacc[i][0]);
            cacc[i][1] = wmma_f16(a, b1, cacc[i][1]);
        }
        if (it + 1 < NIT) fill_store(buf ^ 1);
    }

    #pragma unroll
    for (int i = 0; i < 4; ++i) {
        #pragma unroll
        for (int j = 0; j < 2; ++j) {
            const int mb = mblk * 128 + wm * 64 + i * 16 + ((lane >> 4) << 3);
            const int n  = nblk * 128 + wn * 32 + j * 16 + (lane & 15);
            const float sd = dscale[n] * Q_STEP;
            #pragma unroll
            for (int e = 0; e < 8; ++e) {
                const int m = mb + e;
                float y = cacc[i][j][e] * sd;
                const float* ph = Ph + (size_t)m * RNK;
                float lr = 0.f;
                for (int r = 0; r < RNK; ++r)
                    lr += ph[r] * dB[(size_t)r * HID + n];
                out[(size_t)m * HID + n] = y + LR_SCALE * lr;
            }
        }
    }
}

// ---------------------------------------------------------------- launch
extern "C" void kernel_launch(void* const* d_in, const int* in_sizes, int n_in,
                              void* d_out, int out_size, void* d_ws, size_t ws_size,
                              hipStream_t stream) {
    const float* x  = (const float*)d_in[0];
    const int*   gq = (const int*)  d_in[1];
    const float* gs = (const float*)d_in[2];
    const float* gA = (const float*)d_in[3];
    const float* gB = (const float*)d_in[4];
    const int*   uq = (const int*)  d_in[5];
    const float* us = (const float*)d_in[6];
    const float* uA = (const float*)d_in[7];
    const float* uB = (const float*)d_in[8];
    const int*   dq = (const int*)  d_in[9];
    const float* ds = (const float*)d_in[10];
    const float* dA = (const float*)d_in[11];
    const float* dB = (const float*)d_in[12];
    float* out = (float*)d_out;

    char* ws = (char*)d_ws;
    constexpr size_t XH_BYTES = (size_t)MTOK * HID * 2;   // 33.5 MB
    constexpr size_t HB_BYTES = (size_t)MTOK * IMD * 2;   // 90.2 MB
    constexpr size_t P_BYTES  = (size_t)MTOK * RNK * 4;   // 0.5 MB each
    _Float16* xh   = (_Float16*)(ws);
    _Float16* hbuf = (_Float16*)(ws + XH_BYTES);
    float*    Pg   = (float*)(ws + XH_BYTES + HB_BYTES);
    float*    Pu   = (float*)(ws + XH_BYTES + HB_BYTES + P_BYTES);
    float*    Ph   = (float*)(ws + XH_BYTES + HB_BYTES + 2 * P_BYTES);

    prep_kernel<<<MTOK, 256, 0, stream>>>(x, gA, uA, xh, Pg, Pu);
    gateup_kernel<<<dim3(IMD / 128, MTOK / 128), 256, 0, stream>>>(
        xh, gq, uq, gs, us, gB, uB, Pg, Pu, hbuf);
    projh_kernel<<<MTOK, 256, 0, stream>>>(hbuf, dA, Ph);
    down_kernel<<<dim3(HID / 128, MTOK / 128), 256, 0, stream>>>(
        hbuf, dq, ds, dB, Ph, out);
}
